// SelfAttention_25941602468044
// MI455X (gfx1250) — matrix operand order repaired and checked by reading, not compiled, not run on hardware
//
#include <hip/hip_runtime.h>
#include <hip/hip_bf16.h>

// ---------------------------------------------------------------------------
// CDNA5 (gfx1250) self-attention, bf16 WMMA path with f32 accumulation.
// All matmuls run through v_wmma_f32_16x16x32_bf16.
// ---------------------------------------------------------------------------

typedef unsigned short u16;
typedef __attribute__((ext_vector_type(16))) __bf16 v16bf;
typedef __attribute__((ext_vector_type(8)))  float  v8f;

union Frag {
    v16bf v;
    uint4 q[2];
};

__device__ __forceinline__ u16 f32_to_bf16(float f) {
    unsigned int u = __float_as_uint(f);
    u += 0x7fffu + ((u >> 16) & 1u);   // round-to-nearest-even
    return (u16)(u >> 16);
}

// ---------------------------------------------------------------------------
// Elementwise f32 -> bf16 (vectorized by 4)
// ---------------------------------------------------------------------------
__global__ __launch_bounds__(256) void cvt_f32_bf16_v4(
    const float* __restrict__ in, u16* __restrict__ out, size_t n4)
{
    size_t i = (size_t)blockIdx.x * 256 + threadIdx.x;
    if (i >= n4) return;
    float4 f = ((const float4*)in)[i];
    uint2 o;
    o.x = (unsigned)f32_to_bf16(f.x) | ((unsigned)f32_to_bf16(f.y) << 16);
    o.y = (unsigned)f32_to_bf16(f.z) | ((unsigned)f32_to_bf16(f.w) << 16);
    ((uint2*)out)[i] = o;
}

// ---------------------------------------------------------------------------
// W [K,N] f32 -> WT [N,K] bf16 (32x32 LDS tile transpose)
// ---------------------------------------------------------------------------
__global__ __launch_bounds__(256) void transpose_cvt_f32_bf16(
    const float* __restrict__ W, u16* __restrict__ WT, int K, int N)
{
    __shared__ u16 tile[32][33];
    int x  = blockIdx.x * 32 + threadIdx.x;   // n
    int y0 = blockIdx.y * 32;                 // k
    for (int yy = threadIdx.y; yy < 32; yy += 8)
        tile[yy][threadIdx.x] = f32_to_bf16(W[(size_t)(y0 + yy) * N + x]);
    __syncthreads();
    int xo = blockIdx.y * 32 + threadIdx.x;   // k
    int n0 = blockIdx.x * 32;
    for (int yy = threadIdx.y; yy < 32; yy += 8)
        WT[(size_t)(n0 + yy) * K + xo] = tile[threadIdx.x][yy];
}

// ---------------------------------------------------------------------------
// Batched bf16 transpose: in [B][S][H] -> out [B][H][S]
// ---------------------------------------------------------------------------
__global__ __launch_bounds__(256) void transpose_bf16_batched(
    const u16* __restrict__ in, u16* __restrict__ out, int S, int H)
{
    __shared__ u16 tile[32][33];
    in  += (size_t)blockIdx.z * S * H;
    out += (size_t)blockIdx.z * S * H;
    int h  = blockIdx.x * 32 + threadIdx.x;
    int s0 = blockIdx.y * 32;
    for (int yy = threadIdx.y; yy < 32; yy += 8)
        tile[yy][threadIdx.x] = in[(size_t)(s0 + yy) * H + h];
    __syncthreads();
    int s  = blockIdx.y * 32 + threadIdx.x;
    int h0 = blockIdx.x * 32;
    for (int yy = threadIdx.y; yy < 32; yy += 8)
        out[(size_t)(h0 + yy) * S + s] = tile[threadIdx.x][yy];
}

// ---------------------------------------------------------------------------
// WMMA GEMM:  C[M,N] = A[M,K](bf16, row-major) * B, with B supplied
// pre-transposed as BT[N,K] (bf16, row-major).  C is f32 or bf16.
//
// Block: 256 threads = 8 waves. Block tile 256(M) x 64(N); each wave owns a
// 32x64 C tile (2x4 WMMA accumulators = 64 VGPRs).
//
// K advances in steps of 32 with a global->register->LDS double buffer.
// The next tile is prefetched into 5 individually named uint4 VGPR quads
// (scalars, never an array: arrays captured by lambdas were demoted to
// scratch by the compiler).  The prefetch has no consumer in its own block,
// so the 5 global_load_b128 issue as one clause and their latency is hidden
// across a full compute phase; they are dumped to the single 25KB LDS tile
// at the top of the next iteration.
//
// Fragment gathers (A and B share one contiguous k order, k = 16h + s; C per ISA):
//   A 16x32 bf16 : lane l: row = l&15; elems 0..7  -> K = 16*(l>>4)+0..7
//                                      elems 8..15 -> K = 16*(l>>4)+8..15
//   B 32x16 bf16 : lane l: col = l&15; elems 0..15 -> K = 16*(l>>4)+0..15
//   C 16x16 f32  : vgpr r, lane l: row = r + 8*(l>>4), col = l&15
// ---------------------------------------------------------------------------
#define BLK_M   256
#define BLK_N   64
#define WAVE_M  32
#define LDSTR   40            // padded row stride (elements), 80B = 16B-aligned

__global__ __launch_bounds__(256) void gemm_bf16_wmma(
    const u16* __restrict__ A, const u16* __restrict__ BT, void* __restrict__ Cv,
    int M, int N, int K,
    long long sA, long long sBT, long long sC, int outBf16)
{
    (void)M;
    A  += (size_t)blockIdx.z * sA;
    BT += (size_t)blockIdx.z * sBT;

    __shared__ u16 Alds[BLK_M * LDSTR];   // 20KB
    __shared__ u16 Blds[BLK_N * LDSTR];   //  5KB

    const int tid  = threadIdx.x;
    const int lane = tid & 31;
    const int wave = tid >> 5;
    const int r15  = lane & 15;
    const int h    = lane >> 4;

    const int blockM = blockIdx.y * BLK_M;
    const int blockN = blockIdx.x * BLK_N;

    // Per-thread staging chunk coordinates (fixed for the whole kernel).
    const int arow0 = tid >> 2, ac4 = tid & 3;       // + 64*s rows, s = 0..3
    const int brow  = tid >> 2, bc4 = tid & 3;

    // Base pointers for staging (advance by 32 elements per K-step).
    const u16* Ap = A  + (size_t)(blockM + arow0) * K + ac4 * 8;
    const u16* Bp = BT + (size_t)(blockN + brow)  * K + bc4 * 8;
    const size_t rstep = (size_t)64 * K;

    // LDS staging destinations (bytes are fine as element indices here).
    u16* la0 = &Alds[(arow0      ) * LDSTR + ac4 * 8];
    u16* la1 = &Alds[(arow0 +  64) * LDSTR + ac4 * 8];
    u16* la2 = &Alds[(arow0 + 128) * LDSTR + ac4 * 8];
    u16* la3 = &Alds[(arow0 + 192) * LDSTR + ac4 * 8];
    u16* lb  = &Blds[brow * LDSTR + bc4 * 8];

    v8f acc[2][4] = {};

    // ---- prologue prefetch (kk = 0) into named VGPR quads
    uint4 ra0 = *(const uint4*)(Ap);
    uint4 ra1 = *(const uint4*)(Ap + rstep);
    uint4 ra2 = *(const uint4*)(Ap + 2 * rstep);
    uint4 ra3 = *(const uint4*)(Ap + 3 * rstep);
    uint4 rb  = *(const uint4*)(Bp);

    const int nk = K >> 5;
    for (int kt = 0; kt < nk; ++kt) {
        __syncthreads();                 // all waves done reading previous tile
        // ---- dump prefetched registers to LDS
        *(uint4*)la0 = ra0;
        *(uint4*)la1 = ra1;
        *(uint4*)la2 = ra2;
        *(uint4*)la3 = ra3;
        *(uint4*)lb  = rb;
        // ---- prefetch next tile (no consumer here -> batched, latency
        //      hidden across the whole compute phase below)
        if (kt + 1 < nk) {
            int kk = (kt + 1) << 5;
            ra0 = *(const uint4*)(Ap + kk);
            ra1 = *(const uint4*)(Ap + kk + rstep);
            ra2 = *(const uint4*)(Ap + kk + 2 * rstep);
            ra3 = *(const uint4*)(Ap + kk + 3 * rstep);
            rb  = *(const uint4*)(Bp + kk);
        }
        __syncthreads();                 // LDS tile visible

        Frag bfg[4];
#pragma unroll
        for (int j = 0; j < 4; ++j) {
            int rw = (j * 16 + r15) * LDSTR + h * 16;
            bfg[j].q[0] = *(const uint4*)&Blds[rw];        // K = 16h + 0..7
            bfg[j].q[1] = *(const uint4*)&Blds[rw + 8];    // K = 16h + 8..15
        }
#pragma unroll
        for (int i = 0; i < 2; ++i) {
            Frag af;
            int rw = (wave * WAVE_M + i * 16 + r15) * LDSTR + h * 16;
            af.q[0] = *(const uint4*)&Alds[rw];            // K = 16h + 0..7
            af.q[1] = *(const uint4*)&Alds[rw + 8];        // K = 16h + 8..15
#pragma unroll
            for (int j = 0; j < 4; ++j)
                acc[i][j] = __builtin_amdgcn_wmma_f32_16x16x32_bf16(
                    false, af.v, false, bfg[j].v,
                    (short)0, acc[i][j], false, false);
        }
    }

    const int mW = blockM + wave * WAVE_M;
    if (outBf16) {
        u16* C = (u16*)Cv + (size_t)blockIdx.z * sC;
#pragma unroll
        for (int i = 0; i < 2; ++i)
#pragma unroll
            for (int j = 0; j < 4; ++j) {
                int gn = blockN + j * 16 + r15;
#pragma unroll
                for (int r = 0; r < 8; ++r) {
                    int gm = mW + i * 16 + 8 * h + r;
                    C[(size_t)gm * N + gn] = f32_to_bf16(acc[i][j][r]);
                }
            }
    } else {
        float* C = (float*)Cv + (size_t)blockIdx.z * sC;
#pragma unroll
        for (int i = 0; i < 2; ++i)
#pragma unroll
            for (int j = 0; j < 4; ++j) {
                int gn = blockN + j * 16 + r15;
#pragma unroll
                for (int r = 0; r < 8; ++r) {
                    int gm = mW + i * 16 + 8 * h + r;
                    C[(size_t)gm * N + gn] = acc[i][j][r];
                }
            }
    }
}

// ---------------------------------------------------------------------------
// Row softmax with folded post-scale:  P = softmax(S_row) * scale  -> bf16
// ---------------------------------------------------------------------------
__global__ __launch_bounds__(256) void softmax_rows(
    const float* __restrict__ S, u16* __restrict__ P, int ncols, float scale)
{
    const size_t row = blockIdx.x;
    const float* s = S + row * (size_t)ncols;
    u16*         p = P + row * (size_t)ncols;
    __shared__ float red[256];
    const int tid = threadIdx.x;

    float m = -3.0e38f;
    for (int c = tid; c < ncols; c += 256) m = fmaxf(m, s[c]);
    red[tid] = m; __syncthreads();
    for (int w = 128; w > 0; w >>= 1) {
        if (tid < w) red[tid] = fmaxf(red[tid], red[tid + w]);
        __syncthreads();
    }
    m = red[0];
    __syncthreads();

    float l = 0.f;
    for (int c = tid; c < ncols; c += 256) l += __expf(s[c] - m);
    red[tid] = l; __syncthreads();
    for (int w = 128; w > 0; w >>= 1) {
        if (tid < w) red[tid] += red[tid + w];
        __syncthreads();
    }
    const float inv = scale / red[0];

    for (int c = tid; c < ncols; c += 256)
        p[c] = f32_to_bf16(__expf(s[c] - m) * inv);
}

// ---------------------------------------------------------------------------
// Host-side orchestration
// ---------------------------------------------------------------------------
extern "C" void kernel_launch(void* const* d_in, const int* in_sizes, int n_in,
                              void* d_out, int out_size, void* d_ws, size_t ws_size,
                              hipStream_t stream)
{
    (void)in_sizes; (void)n_in; (void)out_size; (void)ws_size;

    const float* x  = (const float*)d_in[0];   // [4,2048,1024]
    const float* Wq = (const float*)d_in[1];   // [1024,1024]
    const float* Wk = (const float*)d_in[2];
    const float* Wv = (const float*)d_in[3];
    const float* Wo = (const float*)d_in[4];

    const int B = 4, S = 2048, D = 1024, H = 1024;
    const int BS = B * S;                      // 8192
    const float SCALE = 0.03125f;              // 1024^-0.5

    // Workspace layout (lifetime-aliased, 152 MB total):
    char* ws = (char*)d_ws;
    const size_t MB = 1ull << 20;
    u16*   xbf  = (u16*)(ws +  0 * MB);   // 16MB, dead after v-projection
    u16*   WqT  = (u16*)(ws + 16 * MB);   //  2MB
    u16*   WkT  = (u16*)(ws + 18 * MB);
    u16*   WvT  = (u16*)(ws + 20 * MB);
    u16*   WoT  = (u16*)(ws + 22 * MB);
    u16*   q    = (u16*)(ws + 24 * MB);   // 16MB, dead after scores
    u16*   kbuf = (u16*)(ws + 40 * MB);   // 16MB, dead after scores
    u16*   v    = (u16*)(ws + 56 * MB);   // 16MB, dead after vT
    u16*   vT   = (u16*)(ws + 72 * MB);   // 16MB
    float* Sc   = (float*)(ws + 88 * MB); // 64MB scores
    u16*   P    = q;                      // 32MB, aliases q||k (both dead)
    u16*   attn = xbf;                    // 16MB, aliases xbf (dead)

    dim3 t32x8(32, 8);

    // 1. x -> bf16
    {
        size_t n4 = (size_t)BS * D / 4;
        cvt_f32_bf16_v4<<<(unsigned)((n4 + 255) / 256), 256, 0, stream>>>(x, xbf, n4);
    }
    // 2. weights -> transposed bf16 (BT form for the GEMM kernel)
    transpose_cvt_f32_bf16<<<dim3(32, 32), t32x8, 0, stream>>>(Wq, WqT, D, H);
    transpose_cvt_f32_bf16<<<dim3(32, 32), t32x8, 0, stream>>>(Wk, WkT, D, H);
    transpose_cvt_f32_bf16<<<dim3(32, 32), t32x8, 0, stream>>>(Wv, WvT, D, H);
    transpose_cvt_f32_bf16<<<dim3(32, 32), t32x8, 0, stream>>>(Wo, WoT, H, D);

    // 3. projections: q/k/v = x @ W  (M=8192, N=1024, K=1024), bf16 out
    gemm_bf16_wmma<<<dim3(H / BLK_N, BS / BLK_M, 1), 256, 0, stream>>>(
        xbf, WqT, q,    BS, H, D, 0, 0, 0, 1);
    gemm_bf16_wmma<<<dim3(H / BLK_N, BS / BLK_M, 1), 256, 0, stream>>>(
        xbf, WkT, kbuf, BS, H, D, 0, 0, 0, 1);
    gemm_bf16_wmma<<<dim3(H / BLK_N, BS / BLK_M, 1), 256, 0, stream>>>(
        xbf, WvT, v,    BS, H, D, 0, 0, 0, 1);

    // 4. vT[b][h][s]  (BT operand for P @ v)
    transpose_bf16_batched<<<dim3(H / 32, S / 32, B), t32x8, 0, stream>>>(v, vT, S, H);

    // 5. scores = q @ k^T  (per batch; k itself is the [N,K]=[S,H] BT operand)
    gemm_bf16_wmma<<<dim3(S / BLK_N, S / BLK_M, B), 256, 0, stream>>>(
        q, kbuf, Sc, S, S, H,
        (long long)S * H, (long long)S * H, (long long)S * S, 0);

    // 6. P = softmax(scores) * SCALE  (scale fold is exact: it's linear)
    softmax_rows<<<BS, 256, 0, stream>>>(Sc, P, S, SCALE);

    // 7. attn = P @ v  (K = S = 2048, BT = vT)
    gemm_bf16_wmma<<<dim3(H / BLK_N, S / BLK_M, B), 256, 0, stream>>>(
        P, vT, attn, S, H, S,
        (long long)S * S, (long long)H * S, (long long)S * H, 1);

    // 8. out = attn @ Wo  (f32 straight into d_out)
    gemm_bf16_wmma<<<dim3(D / BLK_N, BS / BLK_M, 1), 256, 0, stream>>>(
        attn, WoT, d_out, BS, D, H, 0, 0, 0, 0);
}
